// Attention_16698832847157
// MI455X (gfx1250) — matrix operand order repaired and checked by reading, not compiled, not run on hardware
//
#include <hip/hip_runtime.h>

typedef __bf16 bf16_t;
typedef __attribute__((ext_vector_type(16))) __bf16 v16bf;
typedef __attribute__((ext_vector_type(8)))  __bf16 v8bf;
typedef __attribute__((ext_vector_type(8)))  float  v8f;

constexpr int SEQ    = 4096;
constexpr int CDIM   = 1024;
constexpr int NHEADS = 16;
constexpr int HD     = 64;
constexpr int NBLK   = 32;   // sequence blocks of 128
constexpr int BLKSZ  = 128;
constexpr int TOPK   = 16;   // 0.5 * NBLK
constexpr int C3     = 3072;

union FragBF { v16bf v; v8bf h[2]; };

static __device__ __forceinline__ v8f wmma_bf16(v16bf a, v16bf b, v8f c) {
  // D = A(16x32 bf16) * B(32x16 bf16) + C(16x16 f32)
  return __builtin_amdgcn_wmma_f32_16x16x32_bf16(false, a, false, b, (short)0, c, false, false);
}

// Low 32 bits of a generic pointer to __shared__ == wave-relative LDS byte
// address (ISA: shared aperture occupies addr[63:32]; the low 32 bits are the LDS address).
static __device__ __forceinline__ unsigned lds_lo(const void* p) {
  return (unsigned)(unsigned long long)(uintptr_t)p;
}

#define ASYNC_WAIT() asm volatile("s_wait_asynccnt 0" ::: "memory")

// ---------------------------------------------------------------- fp32->bf16
__global__ __launch_bounds__(256) void cvt_kernel(const float* __restrict__ src,
                                                  bf16_t* __restrict__ dst, int n4) {
  int i = blockIdx.x * blockDim.x + threadIdx.x;
  if (i < n4) {
    float4 f = ((const float4*)src)[i];
    bf16_t* d = dst + (size_t)i * 4;
    d[0] = (bf16_t)f.x; d[1] = (bf16_t)f.y; d[2] = (bf16_t)f.z; d[3] = (bf16_t)f.w;
  }
}

// ------------------------------------------------- shared 128x128 GEMM core
// C[m0..m0+127][c0..c0+127] = A[m x K] * B[ncols x K]^T  (both bf16 row-major)
// 256 threads = 8 waves arranged 2(m) x 4(n); each wave: 64x32 = 4x2 WMMA tiles.
// Double-buffered LDS filled by async global->LDS DMA (ASYNCcnt); manually
// unrolled x2 so each half uses a compile-time buffer index (keeps WMMA
// accumulators coalesced in place), fully unrolled over K.
template <int K>
static __device__ __forceinline__ void gemm_tile_128x128(
    const bf16_t* __restrict__ A, const bf16_t* __restrict__ B,
    int m0, int c0, v8f acc[4][2]) {
  __shared__ bf16_t As[2][128][40];   // 32 K-chunk + 8 pad
  __shared__ bf16_t Bs[2][128][40];
  const int tid  = threadIdx.x;
  const int w    = tid >> 5, lane = tid & 31;
  const int lr   = lane & 15, lh = lane >> 4;
  const int wm   = w >> 2, wn = w & 3;
  const int r    = tid >> 1, hf = tid & 1;

  for (int mt = 0; mt < 4; ++mt)
    for (int nt = 0; nt < 2; ++nt) acc[mt][nt] = {};

  const bf16_t* ga0 = A + (size_t)(m0 + r) * K + hf * 16;
  const bf16_t* gb0 = B + (size_t)(c0 + r) * K + hf * 16;
  const unsigned la[2] = { lds_lo(&As[0][r][hf * 16]), lds_lo(&As[1][r][hf * 16]) };
  const unsigned lb[2] = { lds_lo(&Bs[0][r][hf * 16]), lds_lo(&Bs[1][r][hf * 16]) };

  auto issue = [&](int k0, int buf) {
    asm volatile(
        "global_load_async_to_lds_b128 %0, %2, off\n\t"
        "global_load_async_to_lds_b128 %0, %2, off offset:16\n\t"
        "global_load_async_to_lds_b128 %1, %3, off\n\t"
        "global_load_async_to_lds_b128 %1, %3, off offset:16"
        :: "v"(la[buf]), "v"(lb[buf]),
           "v"((unsigned long long)(ga0 + k0)),
           "v"((unsigned long long)(gb0 + k0)));
  };

  auto compute = [&](int buf) {
    FragBF af[4], bfg[2];
#pragma unroll
    for (int mt = 0; mt < 4; ++mt) {
      const bf16_t* ap = &As[buf][wm * 64 + mt * 16 + lr][0];
      af[mt].h[0] = *(const v8bf*)(ap + 16 * lh);       // K = 16*lh + 0..7
      af[mt].h[1] = *(const v8bf*)(ap + 16 * lh + 8);   // K = 16*lh + 8..15
    }
#pragma unroll
    for (int nt = 0; nt < 2; ++nt) {
      const bf16_t* bp = &Bs[buf][wn * 32 + nt * 16 + lr][0];
      bfg[nt].h[0] = *(const v8bf*)(bp + 16 * lh);      // K = 16*lh + 0..7
      bfg[nt].h[1] = *(const v8bf*)(bp + 16 * lh + 8);  // K = 16*lh + 8..15
    }
#pragma unroll
    for (int mt = 0; mt < 4; ++mt)
#pragma unroll
      for (int nt = 0; nt < 2; ++nt)
        acc[mt][nt] = wmma_bf16(af[mt].v, bfg[nt].v, acc[mt][nt]);
  };

  constexpr int NIT = K / 32;   // even
  issue(0, 0);
#pragma unroll
  for (int kk = 0; kk < NIT; kk += 2) {
    ASYNC_WAIT();              // buffer 0 writes landed (this wave)
    __syncthreads();           // all waves' writes landed; prior reads retired
    if (kk + 1 < NIT) issue((kk + 1) * 32, 1);
    compute(0);
    ASYNC_WAIT();              // buffer 1 writes landed
    __syncthreads();
    if (kk + 2 < NIT) issue((kk + 2) * 32, 0);
    compute(1);
  }
}

// ----------------------------------------------------------------- QKV GEMM
__global__ __launch_bounds__(256) void qkv_gemm_kernel(
    const bf16_t* __restrict__ xb, const bf16_t* __restrict__ wb,
    bf16_t* __restrict__ Qb, bf16_t* __restrict__ Kb, bf16_t* __restrict__ Vb) {
  const int bx = blockIdx.x;
  const int m0 = (bx % (SEQ / 128)) * 128;
  const int c0 = (bx / (SEQ / 128)) * 128;
  v8f acc[4][2];
  gemm_tile_128x128<CDIM>(xb, wb, m0, c0, acc);

  const int tid = threadIdx.x;
  const int w = tid >> 5, lane = tid & 31, lr = lane & 15, lh = lane >> 4;
  const int wm = w >> 2, wn = w & 3;
  for (int mt = 0; mt < 4; ++mt)
    for (int nt = 0; nt < 2; ++nt)
      for (int i = 0; i < 8; ++i) {
        int row = m0 + wm * 64 + mt * 16 + i + 8 * lh;
        int c   = c0 + wn * 32 + nt * 16 + lr;
        float v = acc[mt][nt][i];
        int which = c >> 10;          // 0=q 1=k 2=v
        int hh    = (c >> 6) & 15;
        int dd    = c & 63;
        if (which == 0) v *= 0.125f;  // fold scale = d^-0.5 into Q
        bf16_t* dst = (which == 0) ? Qb : ((which == 1) ? Kb : Vb);
        dst[((size_t)hh * SEQ + row) * HD + dd] = (bf16_t)v;
      }
}

// ------------------------------------------------------------- block means
__global__ __launch_bounds__(64) void means_kernel(
    const bf16_t* __restrict__ Qb, const bf16_t* __restrict__ Kb,
    float* __restrict__ qm, float* __restrict__ km) {
  int b  = blockIdx.x;        // h*NBLK + blk  (contiguous since SEQ = NBLK*BLKSZ)
  int dd = threadIdx.x;       // 0..63
  const bf16_t* qp = Qb + (size_t)b * BLKSZ * HD + dd;
  const bf16_t* kp = Kb + (size_t)b * BLKSZ * HD + dd;
  float sq = 0.f, sk = 0.f;
  for (int r = 0; r < BLKSZ; ++r) { sq += (float)qp[r * HD]; sk += (float)kp[r * HD]; }
  qm[(size_t)b * HD + dd] = sq * (1.f / BLKSZ);
  km[(size_t)b * HD + dd] = sk * (1.f / BLKSZ);
}

// ----------------------------------------------------- top-k block selection
__global__ __launch_bounds__(32) void select_kernel(
    const float* __restrict__ qm, const float* __restrict__ km,
    unsigned char* __restrict__ mask) {
  int bx = blockIdx.x;                // h*NBLK + qb
  int h  = bx / NBLK;
  int kb = threadIdx.x;               // 0..31
  const float* q = qm + (size_t)bx * HD;
  const float* k = km + (size_t)(h * NBLK + kb) * HD;
  float s = 0.f;
  for (int d = 0; d < HD; ++d) s += q[d] * k[d];
  int cnt = 0;
  for (int j = 0; j < NBLK; ++j) {
    float sj = __shfl(s, j, 32);
    cnt += (sj > s) || (sj == s && j < kb);   // stable rank like lax.top_k
  }
  mask[(size_t)bx * NBLK + kb] = (cnt < TOPK) ? 1 : 0;
}

// -------------------------------------------------- block-sparse attention
__global__ __launch_bounds__(256) void attn_kernel(
    const bf16_t* __restrict__ Qb, const bf16_t* __restrict__ Kb,
    const bf16_t* __restrict__ Vb, const unsigned char* __restrict__ mask,
    bf16_t* __restrict__ Ob) {
  __shared__ bf16_t Ks[BLKSZ][72];    // key tile, row-major [key][d]
  __shared__ bf16_t Vs[HD][136];      // value tile, transposed [d][key]
  __shared__ bf16_t Ps[8][16][40];    // per-wave P chunk [row][key-in-chunk]

  const int bx = blockIdx.x;
  const int h = bx / NBLK, qb = bx % NBLK;
  const int tid = threadIdx.x;
  const int w = tid >> 5, lane = tid & 31, lr = lane & 15, lh = lane >> 4;

  // Q A-fragments for this wave's 16 rows (scale already folded in)
  const bf16_t* qrow = Qb + ((size_t)h * SEQ + qb * BLKSZ + w * 16 + lr) * HD;
  FragBF aq[2];
#pragma unroll
  for (int kc = 0; kc < 2; ++kc) {
    aq[kc].h[0] = *(const v8bf*)(qrow + kc * 32 + 16 * lh);
    aq[kc].h[1] = *(const v8bf*)(qrow + kc * 32 + 16 * lh + 8);
  }

  float mrow[8], lrow[8];
  v8f o[4];
  for (int i = 0; i < 8; ++i) { mrow[i] = -1e30f; lrow[i] = 0.f; }
  for (int t = 0; t < 4; ++t) o[t] = {};

  const unsigned char* mp = mask + (size_t)(h * NBLK + qb) * NBLK;
  for (int kb = 0; kb < NBLK; ++kb) {
    if (!mp[kb]) continue;            // uniform across block -> barrier safe
    __syncthreads();                  // all prior reads of Ks/Vs retired
    {
      const int r = tid >> 1, hf = tid & 1;
      // K tile: async global->LDS DMA, 64B contiguous per thread
      const bf16_t* ksrc = Kb + ((size_t)h * SEQ + kb * BLKSZ + r) * HD + hf * 32;
      unsigned lk = lds_lo(&Ks[r][hf * 32]);
      asm volatile(
          "global_load_async_to_lds_b128 %0, %1, off\n\t"
          "global_load_async_to_lds_b128 %0, %1, off offset:16\n\t"
          "global_load_async_to_lds_b128 %0, %1, off offset:32\n\t"
          "global_load_async_to_lds_b128 %0, %1, off offset:48"
          :: "v"(lk), "v"((unsigned long long)ksrc));
      // V tile: vector loads, transpose scatter into [d][key]
      const bf16_t* vsrc = Vb + ((size_t)h * SEQ + kb * BLKSZ + r) * HD + hf * 32;
      v8bf vv[4];
#pragma unroll
      for (int j = 0; j < 4; ++j) vv[j] = ((const v8bf*)vsrc)[j];
#pragma unroll
      for (int j = 0; j < 4; ++j)
#pragma unroll
        for (int e = 0; e < 8; ++e)
          Vs[hf * 32 + j * 8 + e][r] = vv[j][e];
    }
    ASYNC_WAIT();
    __syncthreads();

#pragma unroll
    for (int c = 0; c < 4; ++c) {     // 32-key chunks
      // ---- S = Q * K^T for 32 keys (2 n-tiles x 2 d-chunks = 4 WMMA)
      v8f s[2];
#pragma unroll
      for (int nt = 0; nt < 2; ++nt) {
        v8f a = {};
#pragma unroll
        for (int kc = 0; kc < 2; ++kc) {
          FragBF bk;
          const bf16_t* kr = &Ks[c * 32 + nt * 16 + lr][kc * 32 + 16 * lh];
          bk.h[0] = ((const v8bf*)kr)[0];
          bk.h[1] = ((const v8bf*)kr)[1];
          a = wmma_bf16(aq[kc].v, bk.v, a);
        }
        s[nt] = a;
      }
      // ---- online softmax (rows spread over 16 lanes per half-wave)
      float alpha[8];
#pragma unroll
      for (int i = 0; i < 8; ++i) {
        float bm = fmaxf(s[0][i], s[1][i]);
        bm = fmaxf(bm, __shfl_xor(bm, 1));
        bm = fmaxf(bm, __shfl_xor(bm, 2));
        bm = fmaxf(bm, __shfl_xor(bm, 4));
        bm = fmaxf(bm, __shfl_xor(bm, 8));
        float nm = fmaxf(mrow[i], bm);
        alpha[i] = __expf(mrow[i] - nm);
        mrow[i]  = nm;
        float p0 = __expf(s[0][i] - nm);
        float p1 = __expf(s[1][i] - nm);
        s[0][i] = p0; s[1][i] = p1;
        float rsum = p0 + p1;
        rsum += __shfl_xor(rsum, 1);
        rsum += __shfl_xor(rsum, 2);
        rsum += __shfl_xor(rsum, 4);
        rsum += __shfl_xor(rsum, 8);
        lrow[i] = lrow[i] * alpha[i] + rsum;
      }
#pragma unroll
      for (int t = 0; t < 4; ++t)
#pragma unroll
        for (int i = 0; i < 8; ++i) o[t][i] *= alpha[i];
      // ---- C-layout -> A-fragment via wave-private LDS transpose
#pragma unroll
      for (int nt = 0; nt < 2; ++nt)
#pragma unroll
        for (int i = 0; i < 8; ++i)
          Ps[w][i + 8 * lh][nt * 16 + lr] = (bf16_t)s[nt][i];
      FragBF ap;
      ap.h[0] = *(const v8bf*)&Ps[w][lr][16 * lh];
      ap.h[1] = *(const v8bf*)&Ps[w][lr][16 * lh + 8];
      // ---- O += P * V (4 d-tiles x 1 k-chunk = 4 WMMA)
#pragma unroll
      for (int t = 0; t < 4; ++t) {
        FragBF bv;
        const bf16_t* vr = &Vs[t * 16 + lr][c * 32 + 16 * lh];
        bv.h[0] = ((const v8bf*)vr)[0];
        bv.h[1] = ((const v8bf*)vr)[1];
        o[t] = wmma_bf16(ap.v, bv.v, o[t]);
      }
    }
  }
  // ---- normalize and write [SEQ][CDIM] with col = h*64 + d
  for (int i = 0; i < 8; ++i) {
    float inv = 1.f / lrow[i];
    int row = qb * BLKSZ + w * 16 + i + 8 * lh;
    for (int t = 0; t < 4; ++t) {
      int col = h * HD + t * 16 + lr;
      Ob[(size_t)row * CDIM + col] = (bf16_t)(o[t][i] * inv);
    }
  }
}

// ------------------------------------------------------------ output proj
__global__ __launch_bounds__(256) void proj_gemm_kernel(
    const bf16_t* __restrict__ Ob, const bf16_t* __restrict__ wpb,
    const float* __restrict__ pbias, float* __restrict__ out) {
  const int bx = blockIdx.x;
  const int m0 = (bx % (SEQ / 128)) * 128;
  const int c0 = (bx / (SEQ / 128)) * 128;
  v8f acc[4][2];
  gemm_tile_128x128<CDIM>(Ob, wpb, m0, c0, acc);

  const int tid = threadIdx.x;
  const int w = tid >> 5, lane = tid & 31, lr = lane & 15, lh = lane >> 4;
  const int wm = w >> 2, wn = w & 3;
  for (int mt = 0; mt < 4; ++mt)
    for (int nt = 0; nt < 2; ++nt)
      for (int i = 0; i < 8; ++i) {
        int row = m0 + wm * 64 + mt * 16 + i + 8 * lh;
        int c   = c0 + wn * 32 + nt * 16 + lr;
        out[(size_t)row * CDIM + c] = acc[mt][nt][i] + pbias[c];
      }
}

// ----------------------------------------------------------------- launcher
extern "C" void kernel_launch(void* const* d_in, const int* in_sizes, int n_in,
                              void* d_out, int out_size, void* d_ws, size_t ws_size,
                              hipStream_t stream) {
  const float* x      = (const float*)d_in[0];
  const float* qkv_w  = (const float*)d_in[1];
  const float* proj_w = (const float*)d_in[2];
  const float* proj_b = (const float*)d_in[3];
  float* out = (float*)d_out;

  char* ws = (char*)d_ws;
  size_t off = 0;
  auto alloc = [&](size_t bytes) -> void* {
    void* p = ws + off;
    off += (bytes + 255) & ~(size_t)255;
    return p;
  };
  bf16_t* xb  = (bf16_t*)alloc((size_t)SEQ * CDIM * 2);
  bf16_t* wqb = (bf16_t*)alloc((size_t)C3 * CDIM * 2);
  bf16_t* wpb = (bf16_t*)alloc((size_t)CDIM * CDIM * 2);
  bf16_t* Qb  = (bf16_t*)alloc((size_t)NHEADS * SEQ * HD * 2);
  bf16_t* Kb  = (bf16_t*)alloc((size_t)NHEADS * SEQ * HD * 2);
  bf16_t* Vb  = (bf16_t*)alloc((size_t)NHEADS * SEQ * HD * 2);
  bf16_t* Ob  = (bf16_t*)alloc((size_t)SEQ * CDIM * 2);
  float*  qm  = (float*)alloc((size_t)NHEADS * NBLK * HD * 4);
  float*  km  = (float*)alloc((size_t)NHEADS * NBLK * HD * 4);
  unsigned char* msk = (unsigned char*)alloc((size_t)NHEADS * NBLK * NBLK);

  cvt_kernel<<<SEQ * CDIM / 4 / 256, 256, 0, stream>>>(x, xb, SEQ * CDIM / 4);
  cvt_kernel<<<C3 * CDIM / 4 / 256, 256, 0, stream>>>(qkv_w, wqb, C3 * CDIM / 4);
  cvt_kernel<<<CDIM * CDIM / 4 / 256, 256, 0, stream>>>(proj_w, wpb, CDIM * CDIM / 4);

  qkv_gemm_kernel<<<(SEQ / 128) * (C3 / 128), 256, 0, stream>>>(xb, wqb, Qb, Kb, Vb);
  means_kernel<<<NHEADS * NBLK, 64, 0, stream>>>(Qb, Kb, qm, km);
  select_kernel<<<NHEADS * NBLK, 32, 0, stream>>>(qm, km, msk);
  attn_kernel<<<NHEADS * NBLK, 256, 0, stream>>>(Qb, Kb, Vb, msk, Ob);
  proj_gemm_kernel<<<(SEQ / 128) * (CDIM / 128), 256, 0, stream>>>(Ob, wpb, proj_b, out);
}
